// RecurrentBlock_40948218200603
// MI455X (gfx1250) — compile-verified, matrix operand order checked by reading, not run on hardware
//
#include <hip/hip_runtime.h>
#include <hip/hip_bf16.h>

typedef __bf16 bf16;
typedef __attribute__((ext_vector_type(16))) __bf16 v16bf;
typedef __attribute__((ext_vector_type(8)))  __bf16 v8bf;
typedef __attribute__((ext_vector_type(8)))  float  v8f;
typedef unsigned int u32x4 __attribute__((ext_vector_type(4)));
typedef int          i32x4 __attribute__((ext_vector_type(4)));
typedef int          i32x8 __attribute__((ext_vector_type(8)));

#define EMBED 256
#define HID   1024
#define NEXP  4
#define NTOK  16384   // B*T = 4*4096
#define TSEQ  4096
#define NB    4

// ---------------- workspace layout (bytes) ----------------
static constexpr size_t OFF_QKVT  = 0;                         // 768*256 bf16   = 393216
static constexpr size_t OFF_PROJT = 393216;                    // 256*256 bf16   = 131072
static constexpr size_t OFF_E1T   = 524288;                    // 4*1024*256 bf16= 2097152
static constexpr size_t OFF_E2T   = 2621440;                   // 4*256*1024 bf16= 2097152
static constexpr size_t OFF_HBF   = 4718592;                   // 16384*256 bf16 = 8388608
static constexpr size_t OFF_QKV   = 13107200;                  // 16384*768 f32  = 50331648
static constexpr size_t OFF_VPREF = 63438848;                  // 16384*256 f32  = 16777216
static constexpr size_t OFF_AGATE = 80216064;                  // 16384*256 bf16 = 8388608
static constexpr size_t OFF_X2    = 88604672;                  // 16384*256 f32  = 16777216
static constexpr size_t OFF_H2BF  = 105381888;                 // 16384*256 bf16 = 8388608
static constexpr size_t OFF_GATES = 113770496;                 // 16384*4  f32   = 262144

// ---------------- WMMA fragment helpers ----------------
// 16-bit A-matrix 16x32 layout (ISA 7.12.2): lane%16 = row M, lane/16 selects
// which contiguous K-octet goes in VGPRs 0-3 and 4-7.
__device__ __forceinline__ v16bf load_frag(const bf16* base, int ld, int k, int lane) {
    const bf16* p = base + (size_t)(lane & 15) * ld + k + ((lane >> 4) << 3);
    v8bf lo = *(const v8bf*)p;
    v8bf hi = *(const v8bf*)(p + 16);
    v16bf r;
#pragma unroll
    for (int i = 0; i < 8; ++i) { r[i] = lo[i]; r[i + 8] = hi[i]; }
    return r;
}

__device__ __forceinline__ v8f wmma_bf16(v16bf a, v16bf b, v8f c) {
    return __builtin_amdgcn_wmma_f32_16x16x32_bf16(false, a, false, b, (short)0, c,
                                                   false, false);
}

// ---------------- TDM: 1-D block load Global -> LDS via Tensor Data Mover ----------------
// Builds a D# per ISA ch.8: group0 {count=1, lds_addr, global_addr[56:0], type=2},
// group1 {data_size=4B, tensor_dim0=tile_dim0=ndwords, dim1=1, stride0=ndwords}.
// Issue from ONE wave; that wave waits TENSORcnt, workgroup barrier covers the rest.
__device__ __forceinline__ void tdm_load_1d(void* lds_dst, const void* gsrc, int ndwords) {
#if defined(__gfx1250__) && __has_builtin(__builtin_amdgcn_tensor_load_to_lds)
    unsigned long long ga = (unsigned long long)(uintptr_t)gsrc;
    unsigned int ldsa = (unsigned int)(uintptr_t)lds_dst;   // generic->LDS: low 32 bits
    u32x4 g0;
    g0[0] = 1u;                                             // count=1, user mode
    g0[1] = ldsa;                                           // lds_addr
    g0[2] = (unsigned int)ga;                               // global_addr[31:0]
    g0[3] = (unsigned int)((ga >> 32) & 0x01FFFFFFull) | (2u << 30);  // ga[56:32], type=2
    i32x8 g1;
    g1[0] = (2 << 16);                                      // wg_mask=0, data_size=2 (4B)
    g1[1] = (ndwords & 0xFFFF) << 16;                       // tensor_dim0[15:0]
    g1[2] = ((ndwords >> 16) & 0xFFFF) | (1 << 16);         // tensor_dim0 hi, tensor_dim1=1
    g1[3] = (ndwords & 0xFFFF) << 16;                       // tile_dim0 = ndwords
    g1[4] = 1;                                              // tile_dim1=1, tile_dim2=0
    g1[5] = ndwords;                                        // tensor_dim0_stride lo
    g1[6] = 0;                                              // stride0 hi, stride1 lo
    g1[7] = 0;
    i32x4 z4a = {0, 0, 0, 0}, z4b = {0, 0, 0, 0};
#if __clang_major__ >= 23
    i32x8 z8 = {0, 0, 0, 0, 0, 0, 0, 0};
    __builtin_amdgcn_tensor_load_to_lds(g0, g1, z4a, z4b, z8, 0);
#else
    __builtin_amdgcn_tensor_load_to_lds(g0, g1, z4a, z4b, 0);
#endif
#if __has_builtin(__builtin_amdgcn_s_wait_tensorcnt)
    __builtin_amdgcn_s_wait_tensorcnt(0);
#else
    asm volatile("s_wait_tensorcnt 0x0" ::: "memory");
#endif
#else
    // fallback: vector copy by the calling wave (32 lanes x 16B)
    const uint4* s = (const uint4*)gsrc;
    uint4* d = (uint4*)lds_dst;
    int lane = threadIdx.x & 31;
    for (int i = lane; i < ndwords / 4; i += 32) d[i] = s[i];
#endif
}

// ---------------- weight convert + transpose ----------------
__global__ void k_transpose_bf16(const float* __restrict__ src, bf16* __restrict__ dst,
                                 int R, int C) {
    int idx = blockIdx.x * blockDim.x + threadIdx.x;
    if (idx >= R * C) return;
    int r = idx / C, c = idx % C;
    dst[(size_t)c * R + r] = (bf16)src[(size_t)r * C + c];
}

// ---------------- LayerNorm -> bf16 ----------------
__global__ void k_ln_bf16(const float* __restrict__ x, const float* __restrict__ g,
                          const float* __restrict__ b, bf16* __restrict__ out) {
    int row = blockIdx.x, c = threadIdx.x;
    int lane = c & 31, wave = c >> 5;
    __shared__ float ws[8], ws2[8];
    float v = x[(size_t)row * EMBED + c];
    float s = v, s2 = v * v;
#pragma unroll
    for (int off = 16; off; off >>= 1) { s += __shfl_down(s, off, 32); s2 += __shfl_down(s2, off, 32); }
    if (lane == 0) { ws[wave] = s; ws2[wave] = s2; }
    __syncthreads();
    if (c == 0) {
        float a = 0, a2 = 0;
        for (int i = 0; i < 8; ++i) { a += ws[i]; a2 += ws2[i]; }
        ws[0] = a; ws2[0] = a2;
    }
    __syncthreads();
    float mean = ws[0] * (1.f / EMBED);
    float var  = ws2[0] * (1.f / EMBED) - mean * mean;
    float hn = (v - mean) * rsqrtf(var + 1e-5f) * g[c] + b[c];
    out[(size_t)row * EMBED + c] = (bf16)hn;
}

// ---------------- generic WMMA GEMM: out = A@B^T(+bias)(+residual) ----------------
__global__ void k_gemm_bf16(const bf16* __restrict__ A, const bf16* __restrict__ Bt,
                            const float* __restrict__ bias, const float* __restrict__ residual,
                            float* __restrict__ out, int M, int N, int K) {
    int lane = threadIdx.x & 31, wave = threadIdx.x >> 5;
    int m0 = blockIdx.y * 16;
    int n0 = blockIdx.x * 128 + wave * 16;
    if (n0 >= N || m0 >= M) return;
    const bf16* aB = A + (size_t)m0 * K;
    const bf16* bB = Bt + (size_t)n0 * K;
    v8f acc = {0.f, 0.f, 0.f, 0.f, 0.f, 0.f, 0.f, 0.f};
    for (int k = 0; k < K; k += 32) {
        v16bf a = load_frag(aB, K, k, lane);
        v16bf b = load_frag(bB, K, k, lane);
        if (k + 32 < K)
            __builtin_prefetch(bB + (size_t)(lane & 15) * K + k + 64, 0, 1);
        acc = wmma_bf16(a, b, acc);
    }
    int col = n0 + (lane & 15);
    int rbase = m0 + ((lane >> 4) << 3);
    float bb = bias ? bias[col] : 0.f;
#pragma unroll
    for (int i = 0; i < 8; ++i) {
        int row = rbase + i;
        float v = acc[i] + bb;
        if (residual) v += residual[(size_t)row * N + col];
        out[(size_t)row * N + col] = v;
    }
}

// ---------------- prefix sum of V along T, per (batch, channel) ----------------
__global__ void k_vprefix(const float* __restrict__ qkv, float* __restrict__ vpref) {
    int b = blockIdx.x, c = threadIdx.x;
    const float* src = qkv + ((size_t)b * TSEQ) * 768 + 512 + c;
    float* dst = vpref + ((size_t)b * TSEQ) * EMBED + c;
    float s = 0.f;
    for (int t = 0; t < TSEQ; ++t) {
        s += src[(size_t)t * 768];
        dst[(size_t)t * EMBED] = s;
    }
}

// ---------------- banded softmax attention ----------------
// window [t-16, t]; causal positions below window contribute exp(0)=1 to the
// softmax row and weight 1/denom on v_j  ->  handled via prefix sums of V.
__global__ void k_attn(const float* __restrict__ qkv, const float* __restrict__ vpref,
                       const float* __restrict__ gate, bf16* __restrict__ agate) {
    int t = blockIdx.x, b = blockIdx.y;
    int tid = threadIdx.x, lane = tid & 31, wave = tid >> 5;
    const float* q = qkv + (((size_t)b * TSEQ + t) * 768);
    const float* kb = qkv + ((size_t)b * TSEQ) * 768 + 256;
    const float* vb = qkv + ((size_t)b * TSEQ) * 768 + 512;
    __shared__ float s_sc[32];
    int jlo = (t - 16 > 0) ? t - 16 : 0;
    int nj = t - jlo + 1;              // <= 17
    int outside = t + 1 - nj;          // causal, below window
    for (int jj = wave; jj < nj; jj += 8) {
        const float* kr = kb + (size_t)(jlo + jj) * 768;
        float p = 0.f;
#pragma unroll
        for (int i = 0; i < 8; ++i) { int c = lane + 32 * i; p += q[c] * kr[c]; }
#pragma unroll
        for (int off = 16; off; off >>= 1) p += __shfl_down(p, off, 32);
        if (lane == 0) s_sc[jj] = p * (1.f / 16.f);   // 1/sqrt(256)
    }
    __syncthreads();
    float mmax = (outside > 0) ? 0.f : -1e30f;
    for (int jj = 0; jj < nj; ++jj) mmax = fmaxf(mmax, s_sc[jj]);
    float denom = (float)outside * __expf(-mmax);
    for (int jj = 0; jj < nj; ++jj) denom += __expf(s_sc[jj] - mmax);
    int c = tid;
    float accv = 0.f;
    for (int jj = 0; jj < nj; ++jj)
        accv += __expf(s_sc[jj] - mmax) * vb[(size_t)(jlo + jj) * 768 + c];
    if (outside > 0)
        accv += __expf(-mmax) * vpref[((size_t)b * TSEQ + (t - 17)) * EMBED + c];
    accv = accv / denom * gate[c];
    agate[((size_t)b * TSEQ + t) * EMBED + c] = (bf16)accv;
}

// ---------------- LN2 + gating softmax ----------------
__global__ void k_ln2_gates(const float* __restrict__ x2, const float* __restrict__ g,
                            const float* __restrict__ b, const float* __restrict__ gw,
                            const float* __restrict__ gb, bf16* __restrict__ h2bf,
                            float* __restrict__ gates) {
    int row = blockIdx.x, c = threadIdx.x;
    int lane = c & 31, wave = c >> 5;
    __shared__ float ws[8], ws2[8], h2s[EMBED], logit[4];
    float v = x2[(size_t)row * EMBED + c];
    float s = v, s2 = v * v;
#pragma unroll
    for (int off = 16; off; off >>= 1) { s += __shfl_down(s, off, 32); s2 += __shfl_down(s2, off, 32); }
    if (lane == 0) { ws[wave] = s; ws2[wave] = s2; }
    __syncthreads();
    if (c == 0) {
        float a = 0, a2 = 0;
        for (int i = 0; i < 8; ++i) { a += ws[i]; a2 += ws2[i]; }
        ws[0] = a; ws2[0] = a2;
    }
    __syncthreads();
    float mean = ws[0] * (1.f / EMBED);
    float var  = ws2[0] * (1.f / EMBED) - mean * mean;
    float hn = (v - mean) * rsqrtf(var + 1e-5f) * g[c] + b[c];
    h2s[c] = hn;
    h2bf[(size_t)row * EMBED + c] = (bf16)hn;
    __syncthreads();
    if (wave < NEXP) {
        float p = 0.f;
#pragma unroll
        for (int i = 0; i < 8; ++i) { int cc = lane + 32 * i; p += h2s[cc] * gw[cc * NEXP + wave]; }
#pragma unroll
        for (int off = 16; off; off >>= 1) p += __shfl_down(p, off, 32);
        if (lane == 0) logit[wave] = p + gb[wave];
    }
    __syncthreads();
    if (c == 0) {
        float mx = logit[0];
        for (int i = 1; i < NEXP; ++i) mx = fmaxf(mx, logit[i]);
        float d = 0.f, ev[NEXP];
        for (int i = 0; i < NEXP; ++i) { ev[i] = __expf(logit[i] - mx); d += ev[i]; }
        for (int i = 0; i < NEXP; ++i) gates[(size_t)row * NEXP + i] = ev[i] / d;
    }
}

// ---------------- fused dense-MoE: both GEMMs per expert, hidden lives in LDS ----------------
// 32 tokens per block; LDS: A tile 32x256 bf16 (16KB, staged by TDM) + hidden tile
// 32x1024 bf16 (64KB) = 80KB of the 320KB WGP pool.
__global__ void k_moe(const bf16* __restrict__ h2bf, const bf16* __restrict__ e1T,
                      const bf16* __restrict__ e2T, const float* __restrict__ e_b1,
                      const float* __restrict__ e_b2, const float* __restrict__ gates,
                      const float* __restrict__ x2, float* __restrict__ out) {
    extern __shared__ char smem[];
    bf16* aT   = (bf16*)smem;                    // [32][256]
    bf16* hidT = (bf16*)(smem + 32 * 256 * 2);   // [32][1024]
    int tid = threadIdx.x, lane = tid & 31, wave = tid >> 5;
    int tok0 = blockIdx.x * 32;
    // stage activation tile into LDS via Tensor Data Mover (wave 0 issues + waits)
    if (wave == 0)
        tdm_load_1d(aT, h2bf + (size_t)tok0 * EMBED, (32 * EMBED * 2) / 4);
    __syncthreads();
    int mw = wave & 1, nw = wave >> 1;
    int colL = lane & 15;
    int hi8 = (lane >> 4) << 3;
    v8f accO[4];
#pragma unroll
    for (int j = 0; j < 4; ++j) accO[j] = (v8f){0.f, 0.f, 0.f, 0.f, 0.f, 0.f, 0.f, 0.f};

    for (int e = 0; e < NEXP; ++e) {
        // ---- GEMM1: hid = gelu(h2 @ w1 + b1) -> LDS bf16 ----
        const bf16* B1 = e1T + (size_t)e * HID * EMBED;
        for (int r = 0; r < 16; ++r) {
            int nt = nw + 4 * r;                       // hidden 16-col tile index 0..63
            const bf16* bB = B1 + (size_t)nt * 16 * EMBED;
            v8f acc = {0.f, 0.f, 0.f, 0.f, 0.f, 0.f, 0.f, 0.f};
            for (int k = 0; k < EMBED; k += 32) {
                v16bf a = load_frag(aT + (size_t)mw * 16 * EMBED, EMBED, k, lane);
                v16bf b = load_frag(bB, EMBED, k, lane);
                acc = wmma_bf16(a, b, acc);
            }
            int ncol = nt * 16 + colL;
            float bias = e_b1[e * HID + ncol];
            bf16* dst = hidT + (size_t)(mw * 16) * HID + ncol;
#pragma unroll
            for (int i = 0; i < 8; ++i) {
                float xh = acc[i] + bias;
                float gl = 0.5f * xh * (1.0f + erff(xh * 0.70710678118f));
                dst[(size_t)(i + hi8) * HID] = (bf16)gl;
            }
        }
        __syncthreads();
        // ---- GEMM2: eo = hid @ w2 + b2, gate-scaled accumulate ----
        const bf16* B2 = e2T + (size_t)e * EMBED * HID;
        for (int j = 0; j < 4; ++j) {
            int n0 = nw * 64 + j * 16;
            const bf16* bB = B2 + (size_t)n0 * HID;
            v8f acc = {0.f, 0.f, 0.f, 0.f, 0.f, 0.f, 0.f, 0.f};
            for (int k = 0; k < HID; k += 32) {
                v16bf a = load_frag(hidT + (size_t)mw * 16 * HID, HID, k, lane);
                v16bf b = load_frag(bB, HID, k, lane);
                if (k + 32 < HID)
                    __builtin_prefetch(bB + (size_t)(lane & 15) * HID + k + 64, 0, 1);
                acc = wmma_bf16(a, b, acc);
            }
            float b2 = e_b2[e * EMBED + n0 + colL];
#pragma unroll
            for (int i = 0; i < 8; ++i) {
                int row = mw * 16 + i + hi8;
                float g = gates[(size_t)(tok0 + row) * NEXP + e];
                accO[j][i] += g * (acc[i] + b2);
            }
        }
        __syncthreads();   // protect hidT before next expert overwrites it
    }
    // ---- epilogue: out = x2 + moe_out ----
#pragma unroll
    for (int j = 0; j < 4; ++j) {
        int col = nw * 64 + j * 16 + colL;
#pragma unroll
        for (int i = 0; i < 8; ++i) {
            int row = tok0 + mw * 16 + i + hi8;
            out[(size_t)row * EMBED + col] = x2[(size_t)row * EMBED + col] + accO[j][i];
        }
    }
}

// ---------------- host launcher ----------------
extern "C" void kernel_launch(void* const* d_in, const int* in_sizes, int n_in,
                              void* d_out, int out_size, void* d_ws, size_t ws_size,
                              hipStream_t stream) {
    (void)in_sizes; (void)n_in; (void)out_size; (void)ws_size;
    const float* x         = (const float*)d_in[0];
    const float* ln1_g     = (const float*)d_in[1];
    const float* ln1_b     = (const float*)d_in[2];
    const float* qkv_w     = (const float*)d_in[3];
    const float* qkv_b     = (const float*)d_in[4];
    const float* proj_w    = (const float*)d_in[5];
    const float* proj_b    = (const float*)d_in[6];
    const float* attn_gate = (const float*)d_in[7];
    const float* ln2_g     = (const float*)d_in[8];
    const float* ln2_b     = (const float*)d_in[9];
    const float* gating_w  = (const float*)d_in[10];
    const float* gating_b  = (const float*)d_in[11];
    const float* e_w1      = (const float*)d_in[12];
    const float* e_b1      = (const float*)d_in[13];
    const float* e_w2      = (const float*)d_in[14];
    const float* e_b2      = (const float*)d_in[15];

    char* ws = (char*)d_ws;
    bf16*  qkvT  = (bf16*)(ws + OFF_QKVT);
    bf16*  projT = (bf16*)(ws + OFF_PROJT);
    bf16*  e1T   = (bf16*)(ws + OFF_E1T);
    bf16*  e2T   = (bf16*)(ws + OFF_E2T);
    bf16*  hbf   = (bf16*)(ws + OFF_HBF);
    float* qkvb  = (float*)(ws + OFF_QKV);
    float* vpref = (float*)(ws + OFF_VPREF);
    bf16*  agate = (bf16*)(ws + OFF_AGATE);
    float* x2b   = (float*)(ws + OFF_X2);
    bf16*  h2bf  = (bf16*)(ws + OFF_H2BF);
    float* gatesb= (float*)(ws + OFF_GATES);

    k_transpose_bf16<<<(256 * 768 + 255) / 256, 256, 0, stream>>>(qkv_w, qkvT, 256, 768);
    k_transpose_bf16<<<(256 * 256 + 255) / 256, 256, 0, stream>>>(proj_w, projT, 256, 256);
    for (int e = 0; e < NEXP; ++e) {
        k_transpose_bf16<<<(256 * 1024 + 255) / 256, 256, 0, stream>>>(
            e_w1 + (size_t)e * 256 * 1024, e1T + (size_t)e * 1024 * 256, 256, 1024);
        k_transpose_bf16<<<(1024 * 256 + 255) / 256, 256, 0, stream>>>(
            e_w2 + (size_t)e * 1024 * 256, e2T + (size_t)e * 256 * 1024, 1024, 256);
    }

    k_ln_bf16<<<NTOK, 256, 0, stream>>>(x, ln1_g, ln1_b, hbf);
    k_gemm_bf16<<<dim3(768 / 128, NTOK / 16), 256, 0, stream>>>(
        hbf, qkvT, qkv_b, nullptr, qkvb, NTOK, 768, 256);
    k_vprefix<<<NB, 256, 0, stream>>>(qkvb, vpref);
    k_attn<<<dim3(TSEQ, NB), 256, 0, stream>>>(qkvb, vpref, attn_gate, agate);
    k_gemm_bf16<<<dim3(256 / 128, NTOK / 16), 256, 0, stream>>>(
        agate, projT, proj_b, x, x2b, NTOK, 256, 256);
    k_ln2_gates<<<NTOK, 256, 0, stream>>>(x2b, ln2_g, ln2_b, gating_w, gating_b, h2bf, gatesb);

    static const int kMoeLds = 32 * 256 * 2 + 32 * 1024 * 2;   // 81920
    hipFuncSetAttribute(reinterpret_cast<const void*>(k_moe),
                        hipFuncAttributeMaxDynamicSharedMemorySize, kMoeLds);
    k_moe<<<NTOK / 32, 256, kMoeLds, stream>>>(h2bf, e1T, e2T, e_b1, e_b2, gatesb, x2b,
                                               (float*)d_out);
}
